// GRAPH_AUTOENCODER_47528108098088
// MI455X (gfx1250) — compile-verified, matrix operand order checked by reading, not run on hardware
//
#include <hip/hip_runtime.h>
#include <hip/hip_fp16.h>
#include <math.h>

// ---------------------------------------------------------------------------
// Types for CDNA5 WMMA (wave32, 16x16x32 f16 -> f32)
// ---------------------------------------------------------------------------
typedef _Float16 half_t;
typedef __attribute__((ext_vector_type(16))) _Float16 v16h;
typedef __attribute__((ext_vector_type(8)))  _Float16 v8h;
typedef __attribute__((ext_vector_type(8)))  float    v8f;

// Model constants (match reference)
#define CN  16384   // B*S nodes
#define CF  128
#define CD  256
#define CL  2
#define CH  8
#define CDH 32
#define CB  32
#define CS  512

#define BSTR 40     // padded LDS stride (halves): 20-bank stride, conflict free

// ---------------------------------------------------------------------------
// Utility kernels
// ---------------------------------------------------------------------------
__global__ void k_zero(float* __restrict__ p, long n) {
    for (long i = (long)blockIdx.x * blockDim.x + threadIdx.x; i < n;
         i += (long)gridDim.x * blockDim.x)
        p[i] = 0.0f;
}

__global__ void k_cvt(const float* __restrict__ s, half_t* __restrict__ d, long n) {
    for (long i = (long)blockIdx.x * blockDim.x + threadIdx.x; i < n;
         i += (long)gridDim.x * blockDim.x)
        d[i] = (half_t)s[i];
}

__global__ void k_copy(const float* __restrict__ s, float* __restrict__ d, long n) {
    for (long i = (long)blockIdx.x * blockDim.x + threadIdx.x; i < n;
         i += (long)gridDim.x * blockDim.x)
        d[i] = s[i];
}

// ---------------------------------------------------------------------------
// Strided/batched WMMA GEMM:  C[z] = A[z] * B[z] (+bias) (+gelu)
//   block = 8 waves (dim3(32,8)); each wave computes a 32x32 output tile
//   (2 M-subtiles x 2 N-subtiles -> 4 v_wmma per k-step), waves stacked
//   along M (block tile = 256 x 32).
//   TRANSB=0: B tile staged in LDS in fragment-major layout; each lane
//             fetches its fragment with two ds_load_b128.
//   TRANSB=1: B fragments are row-contiguous in memory (QK^T case) and are
//             loaded directly with 128-bit global loads (no LDS).
//   epi: 0=none, 1=bias, 2=bias+exact GELU.  outHalf: write f16.
// ---------------------------------------------------------------------------
__device__ __forceinline__ float apply_epi(float v, int epi) {
    if (epi == 2) v = 0.5f * v * (1.0f + erff(v * 0.70710678118f));
    return v;
}

template <int TRANSB>
__global__ void k_gemm(const half_t* __restrict__ A, const half_t* __restrict__ B,
                       const float* __restrict__ bias, void* __restrict__ Cv,
                       int M, int N, int K, int lda, int ldb, int ldc,
                       long sA, long sB, long sC, int epi, int outHalf)
{
    __shared__ __align__(16) half_t Bs[32 * BSTR];
    const int lane = threadIdx.x;
    const int wv   = threadIdx.y;
    const int tid  = wv * 32 + lane;
    const long z   = blockIdx.z;
    const long baseA = z * sA, baseB = z * sB, baseC = z * sC;
    const int tileM = (blockIdx.x * 8 + wv) * 32;   // wave's 32-row slab
    const int tileN = blockIdx.y * 32;              // block's 32-col slab

    const int mrow = lane & 15;          // row within subtile / col within subtile
    const int kb   = (lane >> 4) << 3;   // 0 for lanes 0-15, 8 for lanes 16-31
    const int p0   = kb >> 3;            // 0 or 1: which 8-half part

    v8f acc00 = {}, acc01 = {}, acc10 = {}, acc11 = {};

    for (int k0 = 0; k0 < K; k0 += 32) {
        v16h b0, b1;
        if (TRANSB == 0) {
            __syncthreads();
            // cooperative fill: 32(K) x 32(N) tile, fragment-major permuted
            if (tid < 128) {
                int kk = tid >> 2;            // 0..31 (k row)
                int nh = (tid & 3) * 8;       // 0,8,16,24 (n segment)
                v8h row = *(const v8h*)(B + baseB + (long)(k0 + kk) * ldb + tileN + nh);
                int part = kk >> 3, ip = kk & 7;
#pragma unroll
                for (int i = 0; i < 8; ++i)
                    Bs[(nh + i) * BSTR + part * 8 + ip] = row[i];
            }
            __syncthreads();
            v8h b0lo = *(const v8h*)(Bs + mrow * BSTR + p0 * 8);
            v8h b0hi = *(const v8h*)(Bs + mrow * BSTR + (p0 + 2) * 8);
            v8h b1lo = *(const v8h*)(Bs + (16 + mrow) * BSTR + p0 * 8);
            v8h b1hi = *(const v8h*)(Bs + (16 + mrow) * BSTR + (p0 + 2) * 8);
            b0 = __builtin_shufflevector(b0lo, b0hi, 0,1,2,3,4,5,6,7,8,9,10,11,12,13,14,15);
            b1 = __builtin_shufflevector(b1lo, b1hi, 0,1,2,3,4,5,6,7,8,9,10,11,12,13,14,15);
        } else {
            // B^T fragment == same memory pattern as an A fragment
            const half_t* Bp0 = B + baseB + (long)(tileN + mrow) * ldb + k0 + kb;
            const half_t* Bp1 = B + baseB + (long)(tileN + 16 + mrow) * ldb + k0 + kb;
            v8h b0lo = *(const v8h*)(Bp0);
            v8h b0hi = *(const v8h*)(Bp0 + 16);
            v8h b1lo = *(const v8h*)(Bp1);
            v8h b1hi = *(const v8h*)(Bp1 + 16);
            b0 = __builtin_shufflevector(b0lo, b0hi, 0,1,2,3,4,5,6,7,8,9,10,11,12,13,14,15);
            b1 = __builtin_shufflevector(b1lo, b1hi, 0,1,2,3,4,5,6,7,8,9,10,11,12,13,14,15);
        }

        // A fragments for the two M-subtiles (two 128-bit loads each)
        const half_t* Ap0 = A + baseA + (long)(tileM + mrow) * lda + k0 + kb;
        const half_t* Ap1 = A + baseA + (long)(tileM + 16 + mrow) * lda + k0 + kb;
        v8h a0lo = *(const v8h*)(Ap0);
        v8h a0hi = *(const v8h*)(Ap0 + 16);
        v8h a1lo = *(const v8h*)(Ap1);
        v8h a1hi = *(const v8h*)(Ap1 + 16);
        v16h a0 = __builtin_shufflevector(a0lo, a0hi, 0,1,2,3,4,5,6,7,8,9,10,11,12,13,14,15);
        v16h a1 = __builtin_shufflevector(a1lo, a1hi, 0,1,2,3,4,5,6,7,8,9,10,11,12,13,14,15);

        acc00 = __builtin_amdgcn_wmma_f32_16x16x32_f16(false, a0, false, b0, (short)0, acc00, false, false);
        acc01 = __builtin_amdgcn_wmma_f32_16x16x32_f16(false, a0, false, b1, (short)0, acc01, false, false);
        acc10 = __builtin_amdgcn_wmma_f32_16x16x32_f16(false, a1, false, b0, (short)0, acc10, false, false);
        acc11 = __builtin_amdgcn_wmma_f32_16x16x32_f16(false, a1, false, b1, (short)0, acc11, false, false);
    }

    // store: D layout -> VGPR r holds row (sub + r + 8*(lane>=16)), col = (lane&15)
    const int col0 = tileN + mrow;
    const int rofs = (lane >> 4) << 3;
    float bv0 = (bias != nullptr) ? bias[col0] : 0.0f;
    float bv1 = (bias != nullptr) ? bias[col0 + 16] : 0.0f;
#pragma unroll
    for (int r = 0; r < 8; ++r) {
        long r0 = baseC + (long)(tileM + rofs + r) * ldc;
        long r1 = baseC + (long)(tileM + 16 + rofs + r) * ldc;
        float v00 = apply_epi(acc00[r] + bv0, epi);
        float v01 = apply_epi(acc01[r] + bv1, epi);
        float v10 = apply_epi(acc10[r] + bv0, epi);
        float v11 = apply_epi(acc11[r] + bv1, epi);
        if (outHalf) {
            ((half_t*)Cv)[r0 + col0]      = (half_t)v00;
            ((half_t*)Cv)[r0 + col0 + 16] = (half_t)v01;
            ((half_t*)Cv)[r1 + col0]      = (half_t)v10;
            ((half_t*)Cv)[r1 + col0 + 16] = (half_t)v11;
        } else {
            ((float*)Cv)[r0 + col0]      = v00;
            ((float*)Cv)[r0 + col0 + 16] = v01;
            ((float*)Cv)[r1 + col0]      = v10;
            ((float*)Cv)[r1 + col0 + 16] = v11;
        }
    }
}

// ---------------------------------------------------------------------------
// GCN edge pipeline
// ---------------------------------------------------------------------------
__global__ void k_degree(const int* __restrict__ rowE, int E, int Nn,
                         float* __restrict__ deg)
{
    int ET = E + Nn;
    for (long e = (long)blockIdx.x * blockDim.x + threadIdx.x; e < ET;
         e += (long)gridDim.x * blockDim.x) {
        int r = (e < E) ? rowE[e] : (int)(e - E);
        atomicAdd(&deg[r], 1.0f);
    }
}

__global__ void k_edge_norm(const int* __restrict__ rowE, const int* __restrict__ colE,
                            int E, int Nn, const float* __restrict__ deg,
                            float* __restrict__ wbuf, float* __restrict__ degw)
{
    int ET = E + Nn;
    for (long e = (long)blockIdx.x * blockDim.x + threadIdx.x; e < ET;
         e += (long)gridDim.x * blockDim.x) {
        int r, c;
        if (e < E) { r = rowE[e]; c = colE[e]; } else { r = c = (int)(e - E); }
        float dr = deg[r] > 0.0f ? rsqrtf(deg[r]) : 0.0f;
        float dc = deg[c] > 0.0f ? rsqrtf(deg[c]) : 0.0f;
        float nrm = dr * dc;
        wbuf[e] = nrm;
        atomicAdd(&degw[c], nrm);
    }
}

__global__ void k_edge_w(const int* __restrict__ rowE, const int* __restrict__ colE,
                         int E, int Nn, const float* __restrict__ degw,
                         float* __restrict__ wbuf)
{
    int ET = E + Nn;
    for (long e = (long)blockIdx.x * blockDim.x + threadIdx.x; e < ET;
         e += (long)gridDim.x * blockDim.x) {
        int r, c;
        if (e < E) { r = rowE[e]; c = colE[e]; } else { r = c = (int)(e - E); }
        float dwr = degw[r] > 0.0f ? rsqrtf(degw[r]) : 0.0f;
        float dwc = degw[c] > 0.0f ? rsqrtf(degw[c]) : 0.0f;
        wbuf[e] = dwr * wbuf[e] * dwc;
    }
}

// agg[col, :] += w[e] * hW[row, :]   (4 edges/block, 64 lanes/edge, float4 per lane)
__global__ void k_aggregate(const int* __restrict__ rowE, const int* __restrict__ colE,
                            int E, int Nn, const float* __restrict__ w,
                            const float* __restrict__ hW, float* __restrict__ agg)
{
    int e = blockIdx.x * 4 + (threadIdx.x >> 6);
    int t = threadIdx.x & 63;
    int ET = E + Nn;
    if (e >= ET) return;
    int r, c;
    if (e < E) { r = rowE[e]; c = colE[e]; } else { r = c = e - E; }
    float we = w[e];
    const float4 hv = *(const float4*)(hW + (long)r * CD + t * 4);
    float* dst = agg + (long)c * CD + t * 4;
    atomicAdd(dst + 0, we * hv.x);
    atomicAdd(dst + 1, we * hv.y);
    atomicAdd(dst + 2, we * hv.z);
    atomicAdd(dst + 3, we * hv.w);
}

// hb = BN(agg + bc); out = leaky(leaky(hb) + res)
__global__ void k_bn_leaky(const float* __restrict__ agg, const float* __restrict__ res,
                           const float* __restrict__ bc, const float* __restrict__ g,
                           const float* __restrict__ b, const float* __restrict__ m,
                           const float* __restrict__ v, float* __restrict__ out, long n)
{
    for (long i = (long)blockIdx.x * blockDim.x + threadIdx.x; i < n;
         i += (long)gridDim.x * blockDim.x) {
        int f = (int)(i & (CD - 1));
        float hb = (agg[i] + bc[f] - m[f]) * rsqrtf(v[f] + 1e-5f) * g[f] + b[f];
        float t0 = hb > 0.0f ? hb : 0.1f * hb;
        float s  = t0 + res[i];
        out[i]   = s > 0.0f ? s : 0.1f * s;
    }
}

// per-row L2 normalize (in place), one block (256 threads) per row of 256
__global__ void k_rownorm(float* __restrict__ h)
{
    __shared__ float red[256];
    int row = blockIdx.x, t = threadIdx.x;
    long i = (long)row * CD + t;
    float x = h[i];
    red[t] = x * x; __syncthreads();
    for (int s = 128; s > 0; s >>= 1) { if (t < s) red[t] += red[t + s]; __syncthreads(); }
    float nrm = fmaxf(sqrtf(red[0]), 1e-12f);
    h[i] = x / nrm;
}

// out = LayerNorm(x + y) * g + b ; one block per row of 256
__global__ void k_add_ln(const float* __restrict__ x, const float* __restrict__ y,
                         const float* __restrict__ g, const float* __restrict__ b,
                         float* __restrict__ out)
{
    __shared__ float red[256];
    int row = blockIdx.x, t = threadIdx.x;
    long i = (long)row * CD + t;
    float s = x[i] + y[i];
    red[t] = s; __syncthreads();
    for (int k = 128; k > 0; k >>= 1) { if (t < k) red[t] += red[t + k]; __syncthreads(); }
    float mu = red[0] * (1.0f / CD);
    __syncthreads();
    float d = s - mu;
    red[t] = d * d; __syncthreads();
    for (int k = 128; k > 0; k >>= 1) { if (t < k) red[t] += red[t + k]; __syncthreads(); }
    float var = red[0] * (1.0f / CD);
    out[i] = d * rsqrtf(var + 1e-5f) * g[t] + b[t];
}

// softmax over 512 cols (scaled), f32 scores -> f16 probs; block 256, row per block
__global__ void k_softmax(const float* __restrict__ Sc, half_t* __restrict__ P, float scale)
{
    __shared__ float red[256];
    int row = blockIdx.x, t = threadIdx.x;
    const float* sr = Sc + (long)row * CS;
    float v0 = sr[t] * scale, v1 = sr[t + 256] * scale;
    red[t] = fmaxf(v0, v1); __syncthreads();
    for (int s = 128; s > 0; s >>= 1) { if (t < s) red[t] = fmaxf(red[t], red[t + s]); __syncthreads(); }
    float mx = red[0];
    __syncthreads();
    float e0 = __expf(v0 - mx), e1 = __expf(v1 - mx);
    red[t] = e0 + e1; __syncthreads();
    for (int s = 128; s > 0; s >>= 1) { if (t < s) red[t] += red[t + s]; __syncthreads(); }
    float inv = 1.0f / red[0];
    half_t* pr = P + (long)row * CS;
    pr[t]       = (half_t)(e0 * inv);
    pr[t + 256] = (half_t)(e1 * inv);
}

// ---------------------------------------------------------------------------
// Host side
// ---------------------------------------------------------------------------
static inline void gemm(hipStream_t st, const half_t* A, const half_t* B,
                        const float* bias, void* C, int M, int N, int K,
                        int lda, int ldb, int ldc, long sA, long sB, long sC,
                        int gz, int transB, int epi, int outHalf)
{
    dim3 g(M / 256, N / 32, gz), blk(32, 8);
    if (transB)
        k_gemm<1><<<g, blk, 0, st>>>(A, B, bias, C, M, N, K, lda, ldb, ldc,
                                     sA, sB, sC, epi, outHalf);
    else
        k_gemm<0><<<g, blk, 0, st>>>(A, B, bias, C, M, N, K, lda, ldb, ldc,
                                     sA, sB, sC, epi, outHalf);
}

static inline void zero_f(hipStream_t st, float* p, long n)
{
    int blocks = (int)((n + 255) / 256); if (blocks > 16384) blocks = 16384;
    k_zero<<<blocks, 256, 0, st>>>(p, n);
}
static inline void cvt_f(hipStream_t st, const float* s, half_t* d, long n)
{
    int blocks = (int)((n + 255) / 256); if (blocks > 16384) blocks = 16384;
    k_cvt<<<blocks, 256, 0, st>>>(s, d, n);
}

extern "C" void kernel_launch(void* const* d_in, const int* in_sizes, int n_in,
                              void* d_out, int out_size, void* d_ws, size_t ws_size,
                              hipStream_t stream)
{
    const int E  = in_sizes[1] / 2;
    const int ET = E + CN;

    // ---- inputs -----------------------------------------------------------
    const float* x    = (const float*)d_in[0];
    const int*   ei   = (const int*)d_in[1];
    const int*   rowE = ei;
    const int*   colE = ei + E;
    const float* Wc1 = (const float*)d_in[4];  const float* bc1 = (const float*)d_in[5];
    const float* bn1g = (const float*)d_in[6]; const float* bn1b = (const float*)d_in[7];
    const float* bn1m = (const float*)d_in[8]; const float* bn1v = (const float*)d_in[9];
    const float* Wsc = (const float*)d_in[10]; const float* bsc = (const float*)d_in[11];
    const float* Wc2 = (const float*)d_in[12]; const float* bc2 = (const float*)d_in[13];
    const float* bn2g = (const float*)d_in[14]; const float* bn2b = (const float*)d_in[15];
    const float* bn2m = (const float*)d_in[16]; const float* bn2v = (const float*)d_in[17];
    const float* Wq = (const float*)d_in[18]; const float* bq = (const float*)d_in[19];
    const float* Wk = (const float*)d_in[20]; const float* bk = (const float*)d_in[21];
    const float* Wv = (const float*)d_in[22]; const float* bv = (const float*)d_in[23];
    const float* Wo = (const float*)d_in[24]; const float* bo = (const float*)d_in[25];
    const float* Wf1 = (const float*)d_in[26]; const float* bf1 = (const float*)d_in[27];
    const float* Wf2 = (const float*)d_in[28]; const float* bf2 = (const float*)d_in[29];
    const float* ln1g = (const float*)d_in[30]; const float* ln1b = (const float*)d_in[31];
    const float* ln2g = (const float*)d_in[32]; const float* ln2b = (const float*)d_in[33];

    // ---- workspace carve --------------------------------------------------
    char* base = (char*)d_ws; size_t off = 0;
    auto carve = [&](size_t bytes) -> void* {
        void* p = base + off; off = (off + bytes + 255) & ~(size_t)255; return p;
    };
    half_t* act16 = (half_t*)carve((size_t)CN * CD * 2);          // activation f16
    half_t* Q16   = (half_t*)carve((size_t)CN * CD * 2);
    half_t* K16   = (half_t*)carve((size_t)CN * CD * 2);
    half_t* V16   = (half_t*)carve((size_t)CN * CD * 2);
    half_t* ffn16 = (half_t*)carve((size_t)CN * 4 * CD * 2);
    half_t* P16   = (half_t*)carve((size_t)CH * CS * CS * 2);     // per-graph probs
    float*  scores = (float*)carve((size_t)CH * CS * CS * 4);     // per-graph scores
    half_t* wc1_16 = (half_t*)carve((size_t)CF * CD * 2);
    half_t* wsc_16 = (half_t*)carve((size_t)CF * CD * 2);
    half_t* wc2_16 = (half_t*)carve((size_t)CD * CD * 2);
    half_t* wq16 = (half_t*)carve((size_t)CL * CD * CD * 2);
    half_t* wk16 = (half_t*)carve((size_t)CL * CD * CD * 2);
    half_t* wv16 = (half_t*)carve((size_t)CL * CD * CD * 2);
    half_t* wo16 = (half_t*)carve((size_t)CL * CD * CD * 2);
    half_t* wf1_16 = (half_t*)carve((size_t)CL * CD * 4 * CD * 2);
    half_t* wf2_16 = (half_t*)carve((size_t)CL * 4 * CD * CD * 2);
    float* bufGemm = (float*)carve((size_t)CN * CD * 4);
    float* bufRes  = (float*)carve((size_t)CN * CD * 4);
    float* bufAgg  = (float*)carve((size_t)CN * CD * 4);
    float* bufH    = (float*)carve((size_t)CN * CD * 4);
    float* wEdge   = (float*)carve((size_t)ET * 4);
    float* deg     = (float*)carve((size_t)CN * 4);
    float* degw    = (float*)carve((size_t)CN * 4);
    float* bufAttn = bufAgg;   // alias: free after GCN

    // ---- weight conversion (f32 -> f16) -----------------------------------
    cvt_f(stream, Wc1, wc1_16, (long)CF * CD);
    cvt_f(stream, Wsc, wsc_16, (long)CF * CD);
    cvt_f(stream, Wc2, wc2_16, (long)CD * CD);
    cvt_f(stream, Wq, wq16, (long)CL * CD * CD);
    cvt_f(stream, Wk, wk16, (long)CL * CD * CD);
    cvt_f(stream, Wv, wv16, (long)CL * CD * CD);
    cvt_f(stream, Wo, wo16, (long)CL * CD * CD);
    cvt_f(stream, Wf1, wf1_16, (long)CL * CD * 4 * CD);
    cvt_f(stream, Wf2, wf2_16, (long)CL * 4 * CD * CD);

    // ---- GCN edge normalization (shared by both blocks) -------------------
    zero_f(stream, deg, CN);
    zero_f(stream, degw, CN);
    k_degree<<<2048, 256, 0, stream>>>(rowE, E, CN, deg);
    k_edge_norm<<<2048, 256, 0, stream>>>(rowE, colE, E, CN, deg, wEdge, degw);
    k_edge_w<<<2048, 256, 0, stream>>>(rowE, colE, E, CN, degw, wEdge);

    // ---- GCN block 1 ------------------------------------------------------
    cvt_f(stream, x, act16, (long)CN * CF);
    gemm(stream, act16, wc1_16, nullptr, bufGemm, CN, CD, CF, CF, CD, CD,
         0, 0, 0, 1, 0, 0, 0);                                   // hW = x @ Wc1
    gemm(stream, act16, wsc_16, bsc, bufRes, CN, CD, CF, CF, CD, CD,
         0, 0, 0, 1, 0, 1, 0);                                   // res = x @ Wsc + bsc
    zero_f(stream, bufAgg, (long)CN * CD);
    k_aggregate<<<(ET + 3) / 4, 256, 0, stream>>>(rowE, colE, E, CN, wEdge, bufGemm, bufAgg);
    k_bn_leaky<<<16384, 256, 0, stream>>>(bufAgg, bufRes, bc1, bn1g, bn1b, bn1m, bn1v,
                                          bufH, (long)CN * CD);

    // ---- GCN block 2 ------------------------------------------------------
    cvt_f(stream, bufH, act16, (long)CN * CD);
    gemm(stream, act16, wc2_16, nullptr, bufGemm, CN, CD, CD, CD, CD, CD,
         0, 0, 0, 1, 0, 0, 0);
    zero_f(stream, bufAgg, (long)CN * CD);
    k_aggregate<<<(ET + 3) / 4, 256, 0, stream>>>(rowE, colE, E, CN, wEdge, bufGemm, bufAgg);
    k_bn_leaky<<<16384, 256, 0, stream>>>(bufAgg, bufH, bc2, bn2g, bn2b, bn2m, bn2v,
                                          bufRes, (long)CN * CD);
    k_rownorm<<<CN, 256, 0, stream>>>(bufRes);

    float* xb   = bufRes;
    float* xalt = bufH;

    // ---- Transformer layers ----------------------------------------------
    const float invSqrtDH = 0.17677669529663687f; // 1/sqrt(32)
    for (int l = 0; l < CL; ++l) {
        const half_t* wq = wq16 + (long)l * CD * CD;
        const half_t* wk = wk16 + (long)l * CD * CD;
        const half_t* wv = wv16 + (long)l * CD * CD;
        const half_t* wo = wo16 + (long)l * CD * CD;
        const half_t* w1 = wf1_16 + (long)l * CD * 4 * CD;
        const half_t* w2 = wf2_16 + (long)l * 4 * CD * CD;

        cvt_f(stream, xb, act16, (long)CN * CD);
        gemm(stream, act16, wq, bq + l * CD, Q16, CN, CD, CD, CD, CD, CD,
             0, 0, 0, 1, 0, 1, 1);
        gemm(stream, act16, wk, bk + l * CD, K16, CN, CD, CD, CD, CD, CD,
             0, 0, 0, 1, 0, 1, 1);
        gemm(stream, act16, wv, bv + l * CD, V16, CN, CD, CD, CD, CD, CD,
             0, 0, 0, 1, 0, 1, 1);

        for (int b = 0; b < CB; ++b) {
            long nOff = (long)b * CS * CD;
            // scores[h,q,k] = Q[b,q,h,:] . K[b,k,h,:]   (batched over heads)
            gemm(stream, Q16 + nOff, K16 + nOff, nullptr, scores,
                 CS, CS, CDH, CD, CD, CS,
                 (long)CDH, (long)CDH, (long)CS * CS, CH, 1, 0, 0);
            k_softmax<<<CH * CS, 256, 0, stream>>>(scores, P16, invSqrtDH);
            // attn[b,q,h,:] = P[h,q,:] @ V[b,:,h,:]
            gemm(stream, P16, V16 + nOff, nullptr, bufAttn + nOff,
                 CS, CDH, CS, CS, CD, CD,
                 (long)CS * CS, (long)CDH, (long)CDH, CH, 0, 0, 0);
        }

        // out-proj + residual + LN1
        cvt_f(stream, bufAttn, act16, (long)CN * CD);
        gemm(stream, act16, wo, bo + l * CD, bufGemm, CN, CD, CD, CD, CD, CD,
             0, 0, 0, 1, 0, 1, 0);
        k_add_ln<<<CN, 256, 0, stream>>>(xb, bufGemm, ln1g + l * CD, ln1b + l * CD, xalt);
        { float* t = xb; xb = xalt; xalt = t; }

        // FFN: gelu(x @ W1 + b1) @ W2 + b2, residual + LN2
        cvt_f(stream, xb, act16, (long)CN * CD);
        gemm(stream, act16, w1, bf1 + l * 4 * CD, ffn16, CN, 4 * CD, CD, CD, 4 * CD, 4 * CD,
             0, 0, 0, 1, 0, 2, 1);                               // bias + exact GELU -> f16
        gemm(stream, ffn16, w2, bf2 + l * CD, bufGemm, CN, CD, 4 * CD, 4 * CD, CD, CD,
             0, 0, 0, 1, 0, 1, 0);
        k_add_ln<<<CN, 256, 0, stream>>>(xb, bufGemm, ln2g + l * CD, ln2b + l * CD, xalt);
        { float* t = xb; xb = xalt; xalt = t; }
    }

    // ---- output -----------------------------------------------------------
    k_copy<<<16384, 256, 0, stream>>>(xb, (float*)d_out, (long)CN * CD);
}
